// Model_68186900792047
// MI455X (gfx1250) — matrix operand order repaired and checked by reading, not compiled, not run on hardware
//
#include <hip/hip_runtime.h>

#define HD     128   // heads
#define CDIM   512   // kv_lora_rank
#define RDIM   64    // rotary dim
#define DDIM   576   // C + R
#define NSPLIT 8
#define CHUNK  128   // kv positions per streamed chunk

typedef __attribute__((ext_vector_type(16))) __bf16          v16bf;
typedef __attribute__((ext_vector_type(8)))  float           v8f;
typedef __attribute__((ext_vector_type(8)))  unsigned short  v8us;
typedef __attribute__((ext_vector_type(4)))  unsigned short  v4us;
typedef __attribute__((ext_vector_type(2)))  unsigned short  v2us;

union Frag16 { v16bf bf; v8us h[2]; };

__device__ __forceinline__ unsigned short f2bf(float x) {
  union { float f; unsigned u; } v; v.f = x;
  unsigned r = v.u + 0x7FFFu + ((v.u >> 16) & 1u);   // round-to-nearest-even
  return (unsigned short)(r >> 16);
}

// ---- LDS layout (dynamic, ~211 KB) ----
#define OFF_Q   0
#define SZ_Q    (16 * DDIM * 2)            // 18432  : Q bf16 [16 head][576]
#define OFF_KV  (OFF_Q + SZ_Q)
#define SZ_KV   (CHUNK * DDIM * 2)         // 147456 : K chunk bf16 [128 pos][576]; reused as Vt [512 c][128 pos]
#define OFF_SC  (OFF_KV + SZ_KV)
#define SZ_SC   (16 * 512 * 4)             // 32768  : scores f32 [16 head][512 pos]
#define OFF_P   (OFF_SC + SZ_SC)
#define SZ_P    (16 * 512 * 2)             // 16384  : P bf16 [16 head][512 pos]
#define OFF_RED (OFF_P + SZ_P)
#define SZ_RED  (16 * 16 * 4)              // 1024   : partial reductions
#define OFF_ST  (OFF_RED + SZ_RED)
#define SZ_ST   (2 * 16 * 4)               // m, l per head
#define SMEM_BYTES (OFF_ST + SZ_ST)

// ---------------------------------------------------------------------------
// Kernel 1: RoPE the last k_pe token per batch -> d_out tail + workspace
// ---------------------------------------------------------------------------
__global__ void mla_rope_last_kpe(const float* __restrict__ kbuf,
                                  const float* __restrict__ cs_cache,
                                  const int*  __restrict__ positions,
                                  const int*  __restrict__ indptr,
                                  const int*  __restrict__ indices,
                                  const int*  __restrict__ use_rope_p,
                                  float* __restrict__ ws_last,
                                  float* __restrict__ out_kpe)
{
  const int b = blockIdx.x;
  const int t = threadIdx.x;                 // 0..63
  const int kvs = indptr[b];
  const int sl  = indptr[b + 1] - kvs;
  const int tok = indices[kvs + sl - 1];
  const float* kp = kbuf + (long)tok * DDIM + CDIM;
  float v = kp[t];
  if (use_rope_p[0]) {
    const int p = positions[b];
    const int j = t & 31;
    const float c = cs_cache[(long)p * RDIM + j];
    const float s = cs_cache[(long)p * RDIM + 32 + j];
    const float partner = (t < 32) ? -kp[t + 32] : kp[t - 32];
    v = v * c + partner * s;
  }
  ws_last[b * RDIM + t] = v;
  out_kpe[b * RDIM + t] = v;
}

// ---------------------------------------------------------------------------
// Kernel 2: main split attention. grid = (H/16, NSPLIT, B), 256 threads.
// ---------------------------------------------------------------------------
__global__ __launch_bounds__(256)
void mla_decode_split(const float* __restrict__ q,
                      const float* __restrict__ kbuf,
                      const float* __restrict__ vbuf,
                      const float* __restrict__ cs_cache,
                      const int*  __restrict__ positions,
                      const int*  __restrict__ indptr,
                      const int*  __restrict__ indices,
                      const int*  __restrict__ use_rope_p,
                      const float* __restrict__ last_kpe,
                      float* __restrict__ out)
{
  extern __shared__ char smem[];
  unsigned short* Qs = (unsigned short*)(smem + OFF_Q);
  unsigned short* Ks = (unsigned short*)(smem + OFF_KV);
  unsigned short* Vt = (unsigned short*)(smem + OFF_KV);   // reused after phase 1
  float*          Sc = (float*)(smem + OFF_SC);
  unsigned short* Ps = (unsigned short*)(smem + OFF_P);
  float*          Red = (float*)(smem + OFF_RED);
  float*          StatM = (float*)(smem + OFF_ST);
  float*          StatL = StatM + 16;

  const int tid   = threadIdx.x;
  const int wave  = tid >> 5;      // 0..7 (wave32)
  const int lane  = tid & 31;
  const int lhalf = lane >> 4;
  const int l16   = lane & 15;

  const int h0 = blockIdx.x * 16;  // head group base
  const int sp = blockIdx.y;       // kv split
  const int b  = blockIdx.z;
  const int use_rope = use_rope_p[0];

  const int kv_start = indptr[b];
  const int seq_len  = indptr[b + 1] - kv_start;
  const int klps     = (seq_len + NSPLIT - 1) / NSPLIT;
  const int sstart   = sp * klps;
  int count = seq_len - sstart;
  if (count > klps) count = klps;
  if (count < 0)    count = 0;
  const int pos_b = positions[b];
  const float sm_scale = rsqrtf((float)DDIM);

  // ---- load Q tile (16 heads x 576) -> bf16 LDS, rope last 64 features ----
  for (int e = tid; e < 16 * DDIM; e += 256) {
    const int h = e / DDIM, f = e % DDIM;
    const long gi = ((long)(b * HD + h0 + h)) * DDIM + f;
    float v = q[gi];
    if (use_rope && f >= CDIM) {
      const int rho = f - CDIM;
      const int j = rho & 31;
      const float c = cs_cache[(long)pos_b * RDIM + j];
      const float s = cs_cache[(long)pos_b * RDIM + 32 + j];
      const float partner = (rho < 32) ? -q[gi + 32] : q[gi - 32];
      v = v * c + partner * s;
    }
    Qs[e] = f2bf(v);
  }

  // =======================  Phase 1: scores = Q . K^T  =======================
  const int nch = (count + CHUNK - 1) / CHUNK;
  for (int ch = 0; ch < nch; ++ch) {
    __syncthreads();  // prior chunk's WMMA reads of Ks done; Qs visible on iter 0
    const int base = sstart + ch * CHUNK;
    // cooperative K-chunk load: 128 rows x 576 f32 -> bf16 LDS [pos][feat]
    for (int u = tid; u < CHUNK * (DDIM / 4); u += 256) {
      const int row = u / (DDIM / 4);
      const int f0  = (u % (DDIM / 4)) * 4;
      int g = base + row; if (g > seq_len - 1) g = seq_len - 1;
      const int tok = indices[kv_start + g];
      float4 d;
      if (use_rope && f0 >= CDIM && g == seq_len - 1) {
        d = *(const float4*)(last_kpe + b * RDIM + (f0 - CDIM));   // pre-roped
      } else {
        d = *(const float4*)(kbuf + (long)tok * DDIM + f0);
      }
      v4us pk = { f2bf(d.x), f2bf(d.y), f2bf(d.z), f2bf(d.w) };
      *(v4us*)(Ks + row * DDIM + f0) = pk;
    }
    __syncthreads();

    // wave w computes 16 heads x 16 positions, K-loop over 576 (18 WMMAs)
    v8f acc = {};
    for (int kit = 0; kit < DDIM / 32; ++kit) {
      const int k0 = kit * 32;
      Frag16 A, B;
      const unsigned short* qa = Qs + l16 * DDIM + k0 + 8 * lhalf;
      A.h[0] = *(const v8us*)qa;
      A.h[1] = *(const v8us*)(qa + 16);
      const unsigned short* kb = Ks + (wave * 16 + l16) * DDIM + k0 + 8 * lhalf;
      B.h[0] = *(const v8us*)kb;
      B.h[1] = *(const v8us*)(kb + 16);
      acc = __builtin_amdgcn_wmma_f32_16x16x32_bf16(false, A.bf, false, B.bf,
                                                    (short)0, acc, false, false);
    }
    const int colb = ch * CHUNK + wave * 16 + l16;
    #pragma unroll
    for (int r = 0; r < 8; ++r)
      Sc[(8 * lhalf + r) * 512 + colb] = acc[r] * sm_scale;
  }
  __syncthreads();

  // =======================  softmax over the split  =======================
  {
    const int h = tid >> 4, subl = tid & 15;
    float m = -3.402823466e38f;
    for (int i = subl; i < count; i += 16) m = fmaxf(m, Sc[h * 512 + i]);
    Red[h * 16 + subl] = m;
  }
  __syncthreads();
  if ((tid & 15) == 0) {
    const int h = tid >> 4;
    float m = Red[h * 16];
    #pragma unroll
    for (int i = 1; i < 16; ++i) m = fmaxf(m, Red[h * 16 + i]);
    StatM[h] = m;
  }
  __syncthreads();
  {
    const int h = tid >> 4, subl = tid & 15;
    const float m = StatM[h];
    float sum = 0.f;
    for (int i = subl; i < 512; i += 16) {
      float e = 0.f;
      if (i < count) { e = __expf(Sc[h * 512 + i] - m); sum += e; }
      Ps[h * 512 + i] = f2bf(e);
    }
    Red[h * 16 + subl] = sum;
  }
  __syncthreads();
  if ((tid & 15) == 0) {
    const int h = tid >> 4;
    float s = 0.f;
    #pragma unroll
    for (int i = 0; i < 16; ++i) s += Red[h * 16 + i];
    StatL[h] = s;
  }

  // =======================  Phase 2: O = P . V  =======================
  v8f o0 = {}, o1 = {}, o2 = {}, o3 = {};
  for (int ch = 0; ch < nch; ++ch) {
    __syncthreads();  // previous chunk's Vt reads done; first iter: Ps/StatL visible
    const int base = sstart + ch * CHUNK;
    // V chunk 128 pos x 512 C -> transposed bf16 LDS Vt[c][pos]
    for (int u = tid; u < (CHUNK / 2) * (CDIM / 4); u += 256) {
      const int rp = u / (CDIM / 4);
      const int f0 = (u % (CDIM / 4)) * 4;
      const int r0 = rp * 2;
      int g0 = base + r0;     if (g0 > seq_len - 1) g0 = seq_len - 1;
      int g1 = base + r0 + 1; if (g1 > seq_len - 1) g1 = seq_len - 1;
      const int t0 = indices[kv_start + g0];
      const int t1 = indices[kv_start + g1];
      const float4 a  = *(const float4*)(vbuf + (long)t0 * CDIM + f0);
      const float4 bb = *(const float4*)(vbuf + (long)t1 * CDIM + f0);
      const float* af = (const float*)&a;
      const float* bf = (const float*)&bb;
      #pragma unroll
      for (int i = 0; i < 4; ++i) {
        v2us pr = { f2bf(af[i]), f2bf(bf[i]) };
        *(v2us*)(Vt + (f0 + i) * CHUNK + r0) = pr;
      }
    }
    __syncthreads();

    // wave w owns C range [64w, 64w+64): 4 tiles x 4 K-steps of 32 positions
    #pragma unroll
    for (int kk = 0; kk < 4; ++kk) {
      Frag16 A;
      const unsigned short* pa = Ps + l16 * 512 + ch * CHUNK + kk * 32 + 8 * lhalf;
      A.h[0] = *(const v8us*)pa;
      A.h[1] = *(const v8us*)(pa + 16);
      #pragma unroll
      for (int t = 0; t < 4; ++t) {
        Frag16 B;
        const unsigned short* vb =
            Vt + (wave * 64 + t * 16 + l16) * CHUNK + kk * 32 + 8 * lhalf;
        B.h[0] = *(const v8us*)vb;
        B.h[1] = *(const v8us*)(vb + 16);
        v8f& o = (t == 0 ? o0 : t == 1 ? o1 : t == 2 ? o2 : o3);
        o = __builtin_amdgcn_wmma_f32_16x16x32_bf16(false, A.bf, false, B.bf,
                                                    (short)0, o, false, false);
      }
    }
  }

  // =======================  writeback: O/l and lse  =======================
  #pragma unroll
  for (int r = 0; r < 8; ++r) {
    const int hrow = 8 * lhalf + r;
    const float l = StatL[hrow];
    const float inv = (l > 0.f) ? 1.0f / l : 0.f;
    const long obase = (((long)(b * HD + h0 + hrow)) * NSPLIT + sp) * (CDIM + 1);
    out[obase + wave * 64 +  0 + l16] = o0[r] * inv;
    out[obase + wave * 64 + 16 + l16] = o1[r] * inv;
    out[obase + wave * 64 + 32 + l16] = o2[r] * inv;
    out[obase + wave * 64 + 48 + l16] = o3[r] * inv;
  }
  if (tid < 16) {
    const long obase = (((long)(b * HD + h0 + tid)) * NSPLIT + sp) * (CDIM + 1);
    out[obase + CDIM] = StatM[tid] + __logf(StatL[tid]);
  }
}

// ---------------------------------------------------------------------------
extern "C" void kernel_launch(void* const* d_in, const int* in_sizes, int n_in,
                              void* d_out, int out_size, void* d_ws, size_t ws_size,
                              hipStream_t stream) {
  const float* q         = (const float*)d_in[0];
  const float* kbuf      = (const float*)d_in[1];
  const float* vbuf      = (const float*)d_in[2];
  const float* cache     = (const float*)d_in[3];
  const int*   positions = (const int*)d_in[4];
  const int*   indptr    = (const int*)d_in[5];
  const int*   indices   = (const int*)d_in[6];
  const int*   use_rope  = (const int*)d_in[10];

  const int B = in_sizes[0] / (HD * DDIM);
  float* out     = (float*)d_out;
  float* out_kpe = out + (long)B * HD * NSPLIT * (CDIM + 1);
  float* ws_last = (float*)d_ws;   // B * 64 floats

  (void)hipFuncSetAttribute((const void*)mla_decode_split,
                            hipFuncAttributeMaxDynamicSharedMemorySize,
                            SMEM_BYTES);

  mla_rope_last_kpe<<<dim3(B), dim3(RDIM), 0, stream>>>(
      kbuf, cache, positions, indptr, indices, use_rope, ws_last, out_kpe);

  mla_decode_split<<<dim3(HD / 16, NSPLIT, B), dim3(256), SMEM_BYTES, stream>>>(
      q, kbuf, vbuf, cache, positions, indptr, indices, use_rope, ws_last, out);
}
